// Transformer_3393024164287
// MI455X (gfx1250) — matrix operand order repaired and checked by reading, not compiled, not run on hardware
//
#include <hip/hip_runtime.h>
#include <math.h>

// ---------------------------------------------------------------------------
// Types for CDNA5 WMMA (wave32): v16bf A/B fragments, v8f accumulator
// ---------------------------------------------------------------------------
typedef __attribute__((ext_vector_type(16))) __bf16 v16bf;
typedef __attribute__((ext_vector_type(8)))  __bf16 bf16x8;
typedef __attribute__((ext_vector_type(8)))  float  v8f;
typedef __attribute__((ext_vector_type(4)))  unsigned v4u;
typedef __attribute__((ext_vector_type(8)))  int    v8i;
typedef __attribute__((ext_vector_type(4)))  int    v4i;

#define DEPTH 4
#define BATCH 4
#define SEQ   2048
#define DIM   256
#define HEADS 8
#define HDIM  32
#define MLPD  1024
#define MTOT  (BATCH * SEQ)   // 8192 rows

union FragBF { v16bf v; bf16x8 h[2]; __bf16 e[16]; };

#if __has_builtin(__builtin_amdgcn_tensor_load_to_lds)
#define USE_TDM 1
#else
#define USE_TDM 0
#endif

#if __has_builtin(__builtin_amdgcn_permlane16) && __has_builtin(__builtin_amdgcn_update_dpp)
#define FAST_RED 1
#else
#define FAST_RED 0
#endif

// ---------------------------------------------------------------------------
// Cross-lane helpers. Rows of a WMMA C tile live across 16 lanes, so the
// softmax reductions are 16-lane butterflies: xor1/xor2 via DPP quad_perm,
// xor4/xor8 via v_permlane16_b32 (pure VALU, no LDS round trips).
// ---------------------------------------------------------------------------
#if FAST_RED
template <int CTRL>
__device__ inline float dpp_xor(float v) {
  return __int_as_float(
      __builtin_amdgcn_update_dpp(__float_as_int(v), __float_as_int(v), CTRL, 0xF, 0xF, false));
}
__device__ inline float pl16(float v, unsigned lo, unsigned hi) {
  return __int_as_float(
      __builtin_amdgcn_permlane16(__float_as_int(v), __float_as_int(v), (int)lo, (int)hi, false, false));
}
__device__ inline float red16_max(float x) {
  x = fmaxf(x, dpp_xor<0xB1>(x));                      // quad_perm [1,0,3,2]  (xor1)
  x = fmaxf(x, dpp_xor<0x4E>(x));                      // quad_perm [2,3,0,1]  (xor2)
  x = fmaxf(x, pl16(x, 0x32107654u, 0xBA98FEDCu));     // xor4 within 16
  x = fmaxf(x, pl16(x, 0xFEDCBA98u, 0x76543210u));     // xor8 within 16
  return x;
}
__device__ inline float red16_sum(float x) {
  x += dpp_xor<0xB1>(x);
  x += dpp_xor<0x4E>(x);
  x += pl16(x, 0x32107654u, 0xBA98FEDCu);
  x += pl16(x, 0xFEDCBA98u, 0x76543210u);
  return x;
}
#else
__device__ inline float red16_max(float x) {
  for (int off = 8; off > 0; off >>= 1) x = fmaxf(x, __shfl_xor(x, off, 32));
  return x;
}
__device__ inline float red16_sum(float x) {
  for (int off = 8; off > 0; off >>= 1) x += __shfl_xor(x, off, 32);
  return x;
}
#endif

__device__ inline float wave_allred_sum(float x) {
  for (int off = 16; off > 0; off >>= 1) x += __shfl_xor(x, off, 32);
  return x;
}

// ---------------------------------------------------------------------------
// Tensor Data Mover: 2D bf16 tile -> LDS with row padding so the LDS image
// matches the padded layouts the WMMA fragment loads expect.
// ---------------------------------------------------------------------------
#if USE_TDM
__device__ inline void tdm_load_2d_bf16(unsigned lds_off, const __bf16* gptr,
                                        unsigned td0, unsigned td1,
                                        unsigned tile0, unsigned tile1,
                                        unsigned long long stride0,
                                        unsigned pad_int_code, unsigned pad_amt_code) {
  unsigned long long ga = (unsigned long long)(uintptr_t)gptr;
  v4u g0;
  g0[0] = 1u;                                            // count=1, user mode
  g0[1] = lds_off;                                       // LDS byte address
  g0[2] = (unsigned)ga;                                  // global_addr[31:0]
  g0[3] = (unsigned)((ga >> 32) & 0x01FFFFFFu) | (2u << 30); // addr[56:32] | type=2
  v8i g1;
  g1[0] = (int)((1u << 16)                               // data_size = 2 bytes
              | (1u << 20)                               // pad_enable
              | (pad_int_code << 22)                     // pad interval (DWORDs, log code)
              | (pad_amt_code << 25));                   // pad amount (DWORDs-1)
  g1[1] = (int)((td0 & 0xFFFFu) << 16);                  // tensor_dim0[15:0] @ bits 63:48
  g1[2] = (int)((td0 >> 16) | ((td1 & 0xFFFFu) << 16));  // dim0 hi | dim1 lo
  g1[3] = (int)((td1 >> 16) | (tile0 << 16));            // dim1 hi | tile_dim0
  g1[4] = (int)(tile1 & 0xFFFFu);                        // tile_dim1 (tile_dim2 = 0)
  g1[5] = (int)(stride0 & 0xFFFFFFFFull);                // tensor_dim0_stride lo
  g1[6] = (int)((stride0 >> 32) & 0xFFFFull);            // stride hi (dim1_stride = 0)
  g1[7] = 0;
  v4i z4 = {0, 0, 0, 0};
#if __clang_major__ >= 23
  v8i z8 = {0, 0, 0, 0, 0, 0, 0, 0};
  __builtin_amdgcn_tensor_load_to_lds(g0, g1, z4, z4, z8, 0);
#else
  __builtin_amdgcn_tensor_load_to_lds(g0, g1, z4, z4, 0);
#endif
}
#endif

// ---------------------------------------------------------------------------
// Elementwise helpers
// ---------------------------------------------------------------------------
__global__ void cvt_f32_to_bf16(const float* __restrict__ in,
                                __bf16* __restrict__ out, int n) {
  int i = blockIdx.x * blockDim.x + threadIdx.x;
  if (i < n) out[i] = (__bf16)in[i];
}

__global__ void copy_f32(const float* __restrict__ in,
                         float* __restrict__ out, int n) {
  int i = blockIdx.x * blockDim.x + threadIdx.x;
  if (i < n) out[i] = in[i];
}

// ---------------------------------------------------------------------------
// LayerNorm over DIM=256, one wave per row, fp32 in -> bf16 out
// ---------------------------------------------------------------------------
__global__ __launch_bounds__(256)
void layernorm_bf16(const float* __restrict__ x, const float* __restrict__ g,
                    const float* __restrict__ beta, __bf16* __restrict__ out) {
  const int tid  = threadIdx.x;
  const int lane = tid & 31;
  const int w    = tid >> 5;
  const int row  = blockIdx.x * 8 + w;
  const float* xr = x + (size_t)row * DIM;

  float v[8];
  *(float4*)&v[0] = *(const float4*)(xr + lane * 8);
  *(float4*)&v[4] = *(const float4*)(xr + lane * 8 + 4);

  float s = 0.f;
  for (int i = 0; i < 8; i++) s += v[i];
  s = wave_allred_sum(s);
  float mean = s * (1.0f / DIM);

  float q = 0.f;
  for (int i = 0; i < 8; i++) { float d = v[i] - mean; q += d * d; }
  q = wave_allred_sum(q);
  float rstd = rsqrtf(q * (1.0f / DIM) + 1e-5f);

  bf16x8 o;
  for (int i = 0; i < 8; i++) {
    int c = lane * 8 + i;
    o[i] = (__bf16)((v[i] - mean) * rstd * g[c] + beta[c]);
  }
  *(bf16x8*)(out + (size_t)row * DIM + lane * 8) = o;
}

// ---------------------------------------------------------------------------
// bf16 GEMM: C[M,N] = act( A[M,K] @ B[K,N] + bias ) (+ residual)
// Block 256 thr (8 waves), tile 128x128, K-step 32, per wave 2x4 WMMA tiles.
// TDM path: double-buffered LDS tiles delivered by tensor_load_to_lds (wave 0
// issues, TENSORcnt throttles, barrier publishes), compute overlaps DMA.
// ---------------------------------------------------------------------------
template <typename OutT, bool GELU, bool RESID>
__global__ __launch_bounds__(256)
void gemm_bf16(const __bf16* __restrict__ A, const __bf16* __restrict__ Bw,
               const float* __restrict__ bias, OutT* __restrict__ Cout,
               const float* __restrict__ resid, int Kn, int Nn) {
#if USE_TDM
  __shared__ __bf16 As[2][128][40];   // rows padded 32->40 by TDM (16B-aligned rows)
  __shared__ __bf16 Bs[2][32][136];   // rows padded 128->136 by TDM
#else
  __shared__ __bf16 As[1][128][40];
  __shared__ __bf16 Bs[1][32][136];
#endif

  const int tid  = threadIdx.x;
  const int lane = tid & 31, wave = tid >> 5;
  const int half = lane >> 4, ln15 = lane & 15;
  const int m0 = blockIdx.x * 128, n0 = blockIdx.y * 128;
  const int wm = (wave & 3) * 32;   // 4 waves along M
  const int wn = (wave >> 2) * 64;  // 2 waves along N

  v8f acc[2][4];
  for (int i = 0; i < 2; i++)
    for (int j = 0; j < 4; j++) acc[i][j] = {};

  const int ksteps = Kn / 32;

#if USE_TDM
  if (wave == 0) {
    tdm_load_2d_bf16((unsigned)(uintptr_t)&As[0][0][0],
                     A + (size_t)m0 * Kn, (unsigned)Kn, MTOT, 32, 128,
                     (unsigned long long)Kn, 3 /*16 DW*/, 3 /*4 DW pad*/);
    tdm_load_2d_bf16((unsigned)(uintptr_t)&Bs[0][0][0],
                     Bw + n0, (unsigned)Nn, (unsigned)Kn, 128, 32,
                     (unsigned long long)Nn, 5 /*64 DW*/, 3 /*4 DW pad*/);
  }
#endif

  for (int ks = 0; ks < ksteps; ks++) {
#if USE_TDM
    const int buf = ks & 1;
    if (wave == 0) {
      if (ks + 1 < ksteps) {
        tdm_load_2d_bf16((unsigned)(uintptr_t)&As[buf ^ 1][0][0],
                         A + (size_t)m0 * Kn + (ks + 1) * 32, (unsigned)Kn, MTOT, 32, 128,
                         (unsigned long long)Kn, 3, 3);
        tdm_load_2d_bf16((unsigned)(uintptr_t)&Bs[buf ^ 1][0][0],
                         Bw + (size_t)((ks + 1) * 32) * Nn + n0, (unsigned)Nn, (unsigned)Kn, 128, 32,
                         (unsigned long long)Nn, 5, 3);
        __builtin_amdgcn_s_wait_tensorcnt(2);  // current pair complete, next in flight
      } else {
        __builtin_amdgcn_s_wait_tensorcnt(0);
      }
    }
    __syncthreads();
#else
    const int buf = 0;
    const int arow = tid >> 1, acol = (tid & 1) * 16;
    const int brow = tid >> 3, bcol = (tid & 7) * 16;
    __syncthreads();
    {
      const __bf16* ag = A + (size_t)(m0 + arow) * Kn + ks * 32 + acol;
      *(bf16x8*)&As[0][arow][acol]     = *(const bf16x8*)(ag);
      *(bf16x8*)&As[0][arow][acol + 8] = *(const bf16x8*)(ag + 8);
      const __bf16* bg = Bw + (size_t)(ks * 32 + brow) * Nn + n0 + bcol;
      *(bf16x8*)&Bs[0][brow][bcol]     = *(const bf16x8*)(bg);
      *(bf16x8*)&Bs[0][brow][bcol + 8] = *(const bf16x8*)(bg + 8);
    }
    __syncthreads();
#endif

    FragBF af[2], bfg[4];
    for (int t = 0; t < 2; t++) {
      int r = wm + t * 16 + ln15;
      af[t].h[0] = *(const bf16x8*)&As[buf][r][half * 8];
      af[t].h[1] = *(const bf16x8*)&As[buf][r][16 + half * 8];
    }
    for (int t = 0; t < 4; t++) {
      for (int s = 0; s < 16; s++) {
        int kk = (s & 8) * 2 + half * 8 + (s & 7);
        bfg[t].e[s] = Bs[buf][kk][wn + t * 16 + ln15];
      }
    }
    for (int i = 0; i < 2; i++)
      for (int j = 0; j < 4; j++)
        acc[i][j] = __builtin_amdgcn_wmma_f32_16x16x32_bf16(
            false, af[i].v, false, bfg[j].v, (short)0, acc[i][j], false, false);

#if USE_TDM
    __syncthreads();   // all reads of this buffer done before TDM reuses it
#endif
  }

  // Epilogue: C tile layout -> element (M = r + 8*half, N = ln15)
  for (int i = 0; i < 2; i++)
    for (int j = 0; j < 4; j++) {
      int mb = m0 + wm + i * 16 + half * 8;
      int nb = n0 + wn + j * 16 + ln15;
      float bia = bias ? bias[nb] : 0.f;
      for (int r = 0; r < 8; r++) {
        int m = mb + r;
        float v = acc[i][j][r] + bia;
        if (GELU) v = 0.5f * v * (1.0f + erff(v * 0.70710678118f));
        if (RESID) v += resid[(size_t)m * Nn + nb];
        Cout[(size_t)m * Nn + nb] = (OutT)v;
      }
    }
}

// ---------------------------------------------------------------------------
// Flash attention: one wave per 16-query tile per (batch, head).
// qkv layout: [MTOT, 3*DIM], col = part*256 + h*32 + d (part: 0=q,1=k,2=v)
// K chunk transposed at *store* time into LDS as [hd][key]; B-fragments are gathered per element.
// ---------------------------------------------------------------------------
__global__ __launch_bounds__(128)
void attn_flash(const __bf16* __restrict__ qkv, __bf16* __restrict__ out) {
  __shared__ __bf16 KsT[4][32][40];  // per-wave: [hd][key] (K^T), padded rows
  __shared__ __bf16 Pl[4][16][40];   // per-wave: 16 q x 32 k probs

  const int tid  = threadIdx.x;
  const int lane = tid & 31, w = tid >> 5;
  const int half = lane >> 4, ln15 = lane & 15;
  const int h = blockIdx.y, b = blockIdx.z;
  const int q0 = (blockIdx.x * 4 + w) * 16;
  const float scale = 0.0625f;  // DIM^-0.5 (module scales by dim, not head_dim)

  // Q fragment, loaded once, straight from global (rows stride 3*DIM)
  FragBF qf;
  {
    size_t row = (size_t)(b * SEQ + q0 + ln15) * (3 * DIM) + h * HDIM;
    qf.h[0] = *(const bf16x8*)(qkv + row + half * 8);
    qf.h[1] = *(const bf16x8*)(qkv + row + 16 + half * 8);
  }

  float mi[8], li[8];
  for (int r = 0; r < 8; r++) { mi[r] = -3.0e38f; li[r] = 0.f; }
  v8f o0 = {}, o1 = {};

  for (int j0 = 0; j0 < SEQ; j0 += 32) {
    // stage K^T: lane holds key row (j0+lane), scatter to KsT[hd][key=lane]
    {
      size_t krow = (size_t)(b * SEQ + j0 + lane) * (3 * DIM) + DIM + h * HDIM;
      bf16x8 k0 = *(const bf16x8*)(qkv + krow);
      bf16x8 k1 = *(const bf16x8*)(qkv + krow + 8);
      bf16x8 k2 = *(const bf16x8*)(qkv + krow + 16);
      bf16x8 k3 = *(const bf16x8*)(qkv + krow + 24);
      for (int d = 0; d < 8; d++) {
        KsT[w][d][lane]      = k0[d];
        KsT[w][8 + d][lane]  = k1[d];
        KsT[w][16 + d][lane] = k2[d];
        KsT[w][24 + d][lane] = k3[d];
      }
    }
    // B fragments for S = Q·K^T: lane%16 = key, 16 hd per lane in table k order
    FragBF kb0, kb1;
    for (int s = 0; s < 16; s++) {
      int kk = (s & 8) * 2 + half * 8 + (s & 7);
      kb0.e[s] = KsT[w][kk][ln15];
      kb1.e[s] = KsT[w][kk][16 + ln15];
    }

    v8f z = {};
    v8f s0 = __builtin_amdgcn_wmma_f32_16x16x32_bf16(
        false, qf.v, false, kb0.v, (short)0, z, false, false);
    v8f s1 = __builtin_amdgcn_wmma_f32_16x16x32_bf16(
        false, qf.v, false, kb1.v, (short)0, z, false, false);

    // online softmax update per row (row M = r + 8*half across 16 lanes)
    float p0[8], p1[8];
    for (int r = 0; r < 8; r++) {
      float a = s0[r] * scale, c = s1[r] * scale;
      float t  = red16_max(fmaxf(a, c));
      float mn = fmaxf(mi[r], t);
      float e0 = __expf(a - mn), e1 = __expf(c - mn);
      float rs = red16_sum(e0 + e1);
      float al = __expf(mi[r] - mn);
      li[r] = li[r] * al + rs;
      mi[r] = mn;
      o0[r] *= al;
      o1[r] *= al;
      p0[r] = e0;
      p1[r] = e1;
    }
    // restage P into A-layout via LDS (same-wave LDS ops are ordered)
    for (int r = 0; r < 8; r++) {
      Pl[w][half * 8 + r][ln15]      = (__bf16)p0[r];
      Pl[w][half * 8 + r][16 + ln15] = (__bf16)p1[r];
    }
    FragBF pf;
    pf.h[0] = *(const bf16x8*)&Pl[w][ln15][half * 8];
    pf.h[1] = *(const bf16x8*)&Pl[w][ln15][16 + half * 8];

    // V fragments direct from global: lane%16 = hd, 16 keys per lane in table k order
    FragBF vb0, vb1;
    for (int s = 0; s < 16; s++) {
      int kk = (s & 8) * 2 + half * 8 + (s & 7);
      size_t vrow = (size_t)(b * SEQ + j0 + kk) * (3 * DIM) + 2 * DIM + h * HDIM;
      vb0.e[s] = qkv[vrow + ln15];
      vb1.e[s] = qkv[vrow + 16 + ln15];
    }
    o0 = __builtin_amdgcn_wmma_f32_16x16x32_bf16(
        false, pf.v, false, vb0.v, (short)0, o0, false, false);
    o1 = __builtin_amdgcn_wmma_f32_16x16x32_bf16(
        false, pf.v, false, vb1.v, (short)0, o1, false, false);
  }

  // write normalized output: [MTOT, DIM], col = h*32 + d
  for (int r = 0; r < 8; r++) {
    int m = half * 8 + r;
    size_t row = (size_t)(b * SEQ + q0 + m) * DIM + h * HDIM;
    float inv = 1.0f / li[r];
    out[row + ln15]      = (__bf16)(o0[r] * inv);
    out[row + 16 + ln15] = (__bf16)(o1[r] * inv);
  }
}

// ---------------------------------------------------------------------------
// Host-side orchestration
// ---------------------------------------------------------------------------
extern "C" void kernel_launch(void* const* d_in, const int* in_sizes, int n_in,
                              void* d_out, int out_size, void* d_ws,
                              size_t ws_size, hipStream_t stream) {
  (void)in_sizes; (void)n_in; (void)out_size; (void)ws_size;

  const float* x_in  = (const float*)d_in[0];
  const float* ln1_g = (const float*)d_in[1];
  const float* ln1_b = (const float*)d_in[2];
  const float* w_qkv = (const float*)d_in[3];
  const float* w_out = (const float*)d_in[4];
  const float* b_out = (const float*)d_in[5];
  const float* ln2_g = (const float*)d_in[6];
  const float* ln2_b = (const float*)d_in[7];
  const float* w1    = (const float*)d_in[8];
  const float* b1    = (const float*)d_in[9];
  const float* w2    = (const float*)d_in[10];
  const float* b2    = (const float*)d_in[11];

  char* ws = (char*)d_ws;
  size_t off = 0;
  float*  xw    = (float*)(ws + off);  off += (size_t)MTOT * DIM * 4;
  __bf16* xn    = (__bf16*)(ws + off); off += (size_t)MTOT * DIM * 2;
  __bf16* qkvb  = (__bf16*)(ws + off); off += (size_t)MTOT * 3 * DIM * 2;
  __bf16* attnb = (__bf16*)(ws + off); off += (size_t)MTOT * DIM * 2;
  __bf16* hb    = (__bf16*)(ws + off); off += (size_t)MTOT * MLPD * 2;
  __bf16* wqc   = (__bf16*)(ws + off); off += (size_t)DEPTH * DIM * 3 * DIM * 2;
  __bf16* woc   = (__bf16*)(ws + off); off += (size_t)DEPTH * DIM * DIM * 2;
  __bf16* w1c   = (__bf16*)(ws + off); off += (size_t)DEPTH * DIM * MLPD * 2;
  __bf16* w2c   = (__bf16*)(ws + off); off += (size_t)DEPTH * MLPD * DIM * 2;

  auto cvt = [&](const float* src, __bf16* dst, int n) {
    cvt_f32_to_bf16<<<(n + 255) / 256, 256, 0, stream>>>(src, dst, n);
  };
  cvt(w_qkv, wqc, DEPTH * DIM * 3 * DIM);
  cvt(w_out, woc, DEPTH * DIM * DIM);
  cvt(w1,    w1c, DEPTH * DIM * MLPD);
  cvt(w2,    w2c, DEPTH * MLPD * DIM);

  {
    int n = MTOT * DIM;
    copy_f32<<<(n + 255) / 256, 256, 0, stream>>>(x_in, xw, n);
  }

  for (int l = 0; l < DEPTH; l++) {
    const __bf16* wq_l = wqc + (size_t)l * DIM * 3 * DIM;
    const __bf16* wo_l = woc + (size_t)l * DIM * DIM;
    const __bf16* w1_l = w1c + (size_t)l * DIM * MLPD;
    const __bf16* w2_l = w2c + (size_t)l * MLPD * DIM;

    // x + attn(LN1(x))
    layernorm_bf16<<<MTOT / 8, 256, 0, stream>>>(
        xw, ln1_g + l * DIM, ln1_b + l * DIM, xn);
    gemm_bf16<__bf16, false, false><<<dim3(MTOT / 128, (3 * DIM) / 128), 256, 0, stream>>>(
        xn, wq_l, nullptr, qkvb, nullptr, DIM, 3 * DIM);
    attn_flash<<<dim3(SEQ / 64, HEADS, BATCH), 128, 0, stream>>>(qkvb, attnb);
    gemm_bf16<float, false, true><<<dim3(MTOT / 128, DIM / 128), 256, 0, stream>>>(
        attnb, wo_l, b_out + l * DIM, xw, xw, DIM, DIM);

    // x + ffn(LN2(x))
    layernorm_bf16<<<MTOT / 8, 256, 0, stream>>>(
        xw, ln2_g + l * DIM, ln2_b + l * DIM, xn);
    gemm_bf16<__bf16, true, false><<<dim3(MTOT / 128, MLPD / 128), 256, 0, stream>>>(
        xn, w1_l, b1 + l * MLPD, hb, nullptr, DIM, MLPD);
    gemm_bf16<float, false, true><<<dim3(MTOT / 128, DIM / 128), 256, 0, stream>>>(
        hb, w2_l, b2 + l * DIM, xw, xw, MLPD, DIM);
  }

  {
    int n = MTOT * DIM;
    copy_f32<<<(n + 255) / 256, 256, 0, stream>>>(xw, (float*)d_out, n);
  }
}
